// SelfAttentionAISummer_38087769981394
// MI455X (gfx1250) — matrix operand order repaired and checked by reading, not compiled, not run on hardware
//
#include <hip/hip_runtime.h>
#include <hip/hip_bf16.h>

#define DIMD   1024
#define SEQT   2048
#define BATCH  4
#define LDT    40        // LDS row stride in bf16 elements (80 B = 5 x 16 B chunks)

typedef __attribute__((ext_vector_type(16))) __bf16 v16bf;
typedef __attribute__((ext_vector_type(8)))  float  v8f;
typedef __attribute__((ext_vector_type(4)))  int    v4i;

union FragU { unsigned int u[8]; v16bf v; };

#if __has_builtin(__builtin_amdgcn_global_load_async_to_lds_b128) && \
    __has_builtin(__builtin_amdgcn_s_wait_asynccnt)
#define HAVE_ASYNC_LDS 1
__device__ __forceinline__ void async_copy_b128(const void* g, void* l) {
    __builtin_amdgcn_global_load_async_to_lds_b128(
        (__attribute__((address_space(1))) v4i*)(void*)g,
        (__attribute__((address_space(3))) v4i*)l, 0, 0);
}
#else
#define HAVE_ASYNC_LDS 0
#endif

__device__ __forceinline__ unsigned int pack2_bf16(float a, float b) {
    __bf16 ha = (__bf16)a, hb = (__bf16)b;
    unsigned short ua = __builtin_bit_cast(unsigned short, ha);
    unsigned short ub = __builtin_bit_cast(unsigned short, hb);
    return ((unsigned int)ub << 16) | (unsigned int)ua;
}

// A-matrix fragment (16x32 bf16, M x K), tile row-major [m][k] in LDS.
// lanes: m=lane&15, half=lane>>4; VGPR v<4: k=2v+8*half; v>=4: k=16+2(v-4)+8*half
__device__ __forceinline__ v16bf ldsFragA(const __bf16* base) {
    int lane = threadIdx.x & 31;
    int r    = lane & 15;
    int half = lane >> 4;
    const unsigned int* p = (const unsigned int*)(base + r * LDT);
    FragU f;
    int o = half * 4;
#pragma unroll
    for (int v = 0; v < 4; ++v) f.u[v] = p[o + v];
    o = 8 + half * 4;
#pragma unroll
    for (int v = 0; v < 4; ++v) f.u[4 + v] = p[o + v];
    return f.v;
}

// B-matrix fragment (32x16 bf16, K x N), tile stored [n][k] in LDS.
// lane n=lane&15, half=lane>>4; VGPR v<4: k=2v+8*half; v>=4: k=16+2(v-4)+8*half
__device__ __forceinline__ v16bf ldsFragB(const __bf16* base) {
    int lane = threadIdx.x & 31;
    int n    = lane & 15;
    int half = lane >> 4;
    const unsigned int* p = (const unsigned int*)(base + n * LDT);
    FragU f;
#pragma unroll
    for (int v = 0; v < 8; ++v) f.u[v] = p[(v >> 2) * 8 + half * 4 + (v & 3)];
    return f.v;
}

// ---------------------------------------------------------------------------
// Kernel 1: qvk = x @ W^T + b ; de-interleave (d k) -> q,v,k ; q pre-scaled.
// grid (8192/128, 3072/64), block 128 (4 waves). Wave w: rows w*32..w*32+31.
// ---------------------------------------------------------------------------
__global__ __launch_bounds__(128)
void qkv_gemm_kernel(const float* __restrict__ x, const float* __restrict__ W,
                     const float* __restrict__ bias,
                     __bf16* __restrict__ Q, __bf16* __restrict__ Kb,
                     __bf16* __restrict__ V)
{
    __shared__ alignas(16) __bf16 Xs[128 * LDT];
    __shared__ alignas(16) __bf16 Ws[64 * LDT];

    const int m0   = blockIdx.x * 128;
    const int n0   = blockIdx.y * 64;
    const int tid  = threadIdx.x;
    const int wave = tid >> 5;
    const int lane = tid & 31;

    v8f acc[2][4] = {};

    for (int k0 = 0; k0 < DIMD; k0 += 32) {
        __syncthreads();
        // X tile: 128 rows x 32 k -> 2048 float2 pairs; 16 per thread
#pragma unroll
        for (int i = 0; i < 16; ++i) {
            int pi = tid + i * 128;
            int r  = pi >> 4;           // 16 pairs per row
            int c  = (pi & 15) * 2;
            float2 xv = *(const float2*)&x[(size_t)(m0 + r) * DIMD + (k0 + c)];
            *(unsigned int*)&Xs[r * LDT + c] = pack2_bf16(xv.x, xv.y);
        }
        // W tile: 64 rows x 32 k -> 1024 pairs; 8 per thread
#pragma unroll
        for (int i = 0; i < 8; ++i) {
            int pi = tid + i * 128;
            int r  = pi >> 4;
            int c  = (pi & 15) * 2;
            float2 wv = *(const float2*)&W[(size_t)(n0 + r) * DIMD + (k0 + c)];
            *(unsigned int*)&Ws[r * LDT + c] = pack2_bf16(wv.x, wv.y);
        }
        if (k0 + 32 < DIMD)
            __builtin_prefetch(&x[(size_t)(m0 + tid) * DIMD + k0 + 32], 0, 0);
        __syncthreads();

        v16bf a0 = ldsFragA(Xs + (wave * 32)      * LDT);
        v16bf a1 = ldsFragA(Xs + (wave * 32 + 16) * LDT);
#pragma unroll
        for (int nt = 0; nt < 4; ++nt) {
            v16bf b = ldsFragB(Ws + nt * 16 * LDT);
            acc[0][nt] = __builtin_amdgcn_wmma_f32_16x16x32_bf16(
                false, a0, false, b, (short)0, acc[0][nt], false, false);
            acc[1][nt] = __builtin_amdgcn_wmma_f32_16x16x32_bf16(
                false, a1, false, b, (short)0, acc[1][nt], false, false);
        }
    }

    const int half = lane >> 4;
    const int nl   = lane & 15;
    const float qscale = 0.03125f;  // 1024^-0.5
#pragma unroll
    for (int s = 0; s < 2; ++s) {
        int mrow = m0 + wave * 32 + s * 16 + half * 8;
#pragma unroll
        for (int nt = 0; nt < 4; ++nt) {
            int n = n0 + nt * 16 + nl;
            float bv = bias[n];
            int sel = n % 3;
            int j   = n / 3;
#pragma unroll
            for (int r = 0; r < 8; ++r) {
                float cval = acc[s][nt][r] + bv;
                size_t o = (size_t)(mrow + r) * DIMD + j;
                if (sel == 0)      Q[o]  = (__bf16)(cval * qscale);
                else if (sel == 1) V[o]  = (__bf16)cval;
                else               Kb[o] = (__bf16)cval;
            }
        }
    }
}

// ---------------------------------------------------------------------------
// Kernel 2: S[b,i,j] = sum_d Q[b,i,d] * K[b,j,d]   (Q pre-scaled)
// grid (2048/128, 2048/64, 4), block 128. Async B128 staging into LDS.
// ---------------------------------------------------------------------------
__global__ __launch_bounds__(128)
void scores_gemm_kernel(const __bf16* __restrict__ Q, const __bf16* __restrict__ Kb,
                        float* __restrict__ S)
{
    __shared__ alignas(16) __bf16 Qs[128 * LDT];
    __shared__ alignas(16) __bf16 Ks[64 * LDT];

    const int bb   = blockIdx.z;
    const int i0   = blockIdx.x * 128;
    const int j0   = blockIdx.y * 64;
    const int tid  = threadIdx.x;
    const int wave = tid >> 5;
    const int lane = tid & 31;

    const __bf16* Qb = Q  + (size_t)bb * SEQT * DIMD;
    const __bf16* Kp = Kb + (size_t)bb * SEQT * DIMD;

    v8f acc[2][4] = {};

    for (int k0 = 0; k0 < DIMD; k0 += 32) {
        __syncthreads();
#if HAVE_ASYNC_LDS
        // Q tile: 128 rows x 64 B = 512 x B128 chunks; 4 per thread
#pragma unroll
        for (int i = 0; i < 4; ++i) {
            int c = tid + i * 128;
            int r = c >> 2, q = c & 3;
            async_copy_b128(Qb + (size_t)(i0 + r) * DIMD + k0 + q * 8,
                            Qs + r * LDT + q * 8);
        }
        // K tile: 64 rows -> 256 chunks; 2 per thread
#pragma unroll
        for (int i = 0; i < 2; ++i) {
            int c = tid + i * 128;
            int r = c >> 2, q = c & 3;
            async_copy_b128(Kp + (size_t)(j0 + r) * DIMD + k0 + q * 8,
                            Ks + r * LDT + q * 8);
        }
        __builtin_amdgcn_s_wait_asynccnt(0);
#else
#pragma unroll
        for (int i = 0; i < 16; ++i) {
            int idx = tid + i * 128;
            int r = idx >> 4, c2 = idx & 15;
            ((unsigned int*)(Qs + r * LDT))[c2] =
                ((const unsigned int*)(Qb + (size_t)(i0 + r) * DIMD + k0))[c2];
        }
#pragma unroll
        for (int i = 0; i < 8; ++i) {
            int idx = tid + i * 128;
            int r = idx >> 4, c2 = idx & 15;
            ((unsigned int*)(Ks + r * LDT))[c2] =
                ((const unsigned int*)(Kp + (size_t)(j0 + r) * DIMD + k0))[c2];
        }
#endif
        __syncthreads();

        v16bf a0 = ldsFragA(Qs + (wave * 32)      * LDT);
        v16bf a1 = ldsFragA(Qs + (wave * 32 + 16) * LDT);
#pragma unroll
        for (int nt = 0; nt < 4; ++nt) {
            v16bf b = ldsFragB(Ks + nt * 16 * LDT);
            acc[0][nt] = __builtin_amdgcn_wmma_f32_16x16x32_bf16(
                false, a0, false, b, (short)0, acc[0][nt], false, false);
            acc[1][nt] = __builtin_amdgcn_wmma_f32_16x16x32_bf16(
                false, a1, false, b, (short)0, acc[1][nt], false, false);
        }
    }

    const int half = lane >> 4;
    const int nl   = lane & 15;
    float* Srow = S + (size_t)bb * SEQT * SEQT;
#pragma unroll
    for (int s = 0; s < 2; ++s) {
        int mrow = i0 + wave * 32 + s * 16 + half * 8;
#pragma unroll
        for (int nt = 0; nt < 4; ++nt) {
            int j = j0 + nt * 16 + nl;
#pragma unroll
            for (int r = 0; r < 8; ++r)
                Srow[(size_t)(mrow + r) * SEQT + j] = acc[s][nt][r];
        }
    }
}

// ---------------------------------------------------------------------------
// Kernel 3: row softmax over S (2048 cols), write bf16 P. grid 8192, block 256.
// ---------------------------------------------------------------------------
__global__ __launch_bounds__(256)
void softmax_kernel(const float* __restrict__ S, __bf16* __restrict__ P)
{
    __shared__ float red[8];
    const int row = blockIdx.x;
    const float* s = S + (size_t)row * SEQT;
    __bf16* p = P + (size_t)row * SEQT;
    const int tid = threadIdx.x;

    float lmax = -3.402823466e38f;
    for (int c = tid; c < SEQT; c += 256) lmax = fmaxf(lmax, s[c]);
#pragma unroll
    for (int off = 16; off > 0; off >>= 1)
        lmax = fmaxf(lmax, __shfl_xor(lmax, off, 32));
    if ((tid & 31) == 0) red[tid >> 5] = lmax;
    __syncthreads();
    float rmax = red[0];
#pragma unroll
    for (int i = 1; i < 8; ++i) rmax = fmaxf(rmax, red[i]);
    __syncthreads();

    float lsum = 0.0f;
    for (int c = tid; c < SEQT; c += 256) lsum += __expf(s[c] - rmax);
#pragma unroll
    for (int off = 16; off > 0; off >>= 1)
        lsum += __shfl_xor(lsum, off, 32);
    if ((tid & 31) == 0) red[tid >> 5] = lsum;
    __syncthreads();
    float rsum = red[0];
#pragma unroll
    for (int i = 1; i < 8; ++i) rsum += red[i];
    float inv = __frcp_rn(rsum);

    for (int c = tid; c < SEQT; c += 256)
        p[c] = (__bf16)(__expf(s[c] - rmax) * inv);
}

// ---------------------------------------------------------------------------
// Kernel 4: out[b,i,n] = sum_j P[b,i,j] * V[b,j,n]
// grid (2048/128, 1024/64, 4), block 128. P async-staged; V staged transposed.
// ---------------------------------------------------------------------------
__global__ __launch_bounds__(128)
void pv_gemm_kernel(const __bf16* __restrict__ P, const __bf16* __restrict__ V,
                    float* __restrict__ out)
{
    __shared__ alignas(16) __bf16 Ps[128 * LDT];
    __shared__ alignas(16) __bf16 Vs[64 * LDT];   // transposed: Vs[n][j]

    const int bb   = blockIdx.z;
    const int i0   = blockIdx.x * 128;
    const int n0   = blockIdx.y * 64;
    const int tid  = threadIdx.x;
    const int wave = tid >> 5;
    const int lane = tid & 31;

    const __bf16* Pb = P + (size_t)bb * SEQT * SEQT;
    const __bf16* Vb = V + (size_t)bb * SEQT * DIMD;

    v8f acc[2][4] = {};

    for (int j0 = 0; j0 < SEQT; j0 += 32) {
        __syncthreads();
#if HAVE_ASYNC_LDS
        // P tile: 128 rows x 64 B = 512 x B128 chunks; 4 per thread
#pragma unroll
        for (int i = 0; i < 4; ++i) {
            int c = tid + i * 128;
            int r = c >> 2, q = c & 3;
            async_copy_b128(Pb + (size_t)(i0 + r) * SEQT + j0 + q * 8,
                            Ps + r * LDT + q * 8);
        }
#else
#pragma unroll
        for (int i = 0; i < 16; ++i) {
            int idx = tid + i * 128;
            int r = idx >> 4, c2 = idx & 15;
            ((unsigned int*)(Ps + r * LDT))[c2] =
                ((const unsigned int*)(Pb + (size_t)(i0 + r) * SEQT + j0))[c2];
        }
#endif
        // V tile transposed: 32 j x 64 n, coalesced global reads, ds scatter
#pragma unroll
        for (int i = 0; i < 16; ++i) {
            int idx = tid + i * 128;
            int j = idx >> 6;              // 0..31
            int n = idx & 63;
            Vs[n * LDT + j] = Vb[(size_t)(j0 + j) * DIMD + (n0 + n)];
        }
#if HAVE_ASYNC_LDS
        __builtin_amdgcn_s_wait_asynccnt(0);
#endif
        __syncthreads();

        v16bf a0 = ldsFragA(Ps + (wave * 32)      * LDT);
        v16bf a1 = ldsFragA(Ps + (wave * 32 + 16) * LDT);
#pragma unroll
        for (int nt = 0; nt < 4; ++nt) {
            v16bf b = ldsFragB(Vs + nt * 16 * LDT);
            acc[0][nt] = __builtin_amdgcn_wmma_f32_16x16x32_bf16(
                false, a0, false, b, (short)0, acc[0][nt], false, false);
            acc[1][nt] = __builtin_amdgcn_wmma_f32_16x16x32_bf16(
                false, a1, false, b, (short)0, acc[1][nt], false, false);
        }
    }

    const int half = lane >> 4;
    const int nl   = lane & 15;
    float* Ob = out + (size_t)bb * SEQT * DIMD;
#pragma unroll
    for (int s = 0; s < 2; ++s) {
        int mrow = i0 + wave * 32 + s * 16 + half * 8;
#pragma unroll
        for (int nt = 0; nt < 4; ++nt) {
            int n = n0 + nt * 16 + nl;
#pragma unroll
            for (int r = 0; r < 8; ++r)
                Ob[(size_t)(mrow + r) * DIMD + n] = acc[s][nt][r];
        }
    }
}

// ---------------------------------------------------------------------------
extern "C" void kernel_launch(void* const* d_in, const int* in_sizes, int n_in,
                              void* d_out, int out_size, void* d_ws, size_t ws_size,
                              hipStream_t stream)
{
    (void)in_sizes; (void)n_in; (void)out_size; (void)ws_size;

    const float* x    = (const float*)d_in[0];   // [4,2048,1024]
    const float* W    = (const float*)d_in[1];   // [3072,1024]
    const float* bias = (const float*)d_in[2];   // [3072]
    float* out        = (float*)d_out;           // [4,2048,1024]

    char* ws = (char*)d_ws;
    const size_t QKV_BYTES = (size_t)8192 * 1024 * 2;       // 16 MiB each
    const size_t S_BYTES   = (size_t)4 * 2048 * 2048 * 4;   // 64 MiB
    __bf16* Q  = (__bf16*)(ws);
    __bf16* Kb = (__bf16*)(ws + QKV_BYTES);
    __bf16* V  = (__bf16*)(ws + 2 * QKV_BYTES);
    float*  S  = (float*)(ws + 3 * QKV_BYTES);
    __bf16* P  = (__bf16*)(ws + 3 * QKV_BYTES + S_BYTES);

    dim3 blk(128);
    qkv_gemm_kernel<<<dim3(8192 / 128, 3072 / 64), blk, 0, stream>>>(x, W, bias, Q, Kb, V);
    scores_gemm_kernel<<<dim3(SEQT / 128, SEQT / 64, BATCH), blk, 0, stream>>>(Q, Kb, S);
    softmax_kernel<<<dim3(BATCH * SEQT), dim3(256), 0, stream>>>(S, P);
    pv_gemm_kernel<<<dim3(SEQT / 128, DIMD / 64, BATCH), blk, 0, stream>>>(P, V, out);
}
